// MonarchMixerSequenceMixing_52785148068654
// MI455X (gfx1250) — matrix operand order repaired and checked by reading, not compiled, not run on hardware
//
#include <hip/hip_runtime.h>

typedef __attribute__((ext_vector_type(16))) __bf16 v16bf;
typedef __attribute__((ext_vector_type(8)))  __bf16 v8bf;
typedef __attribute__((ext_vector_type(8)))  float  v8f;

static constexpr int Bsz = 4, Lq = 4096, Dm = 768, TD = 2304, BL = 16384;
static constexpr int ORDER = 16;
static constexpr int KREV_L = Lq + 64;   // reversed k with 32-elem zero margins both ends

#define USE_ASYNC_G2L 1

// ---------------------------------------------------------------- converts
__global__ void cvt_f32_to_bf16(const float* __restrict__ s, __bf16* __restrict__ d, int n) {
    int i = blockIdx.x * 256 + threadIdx.x;
    if (i < n) d[i] = (__bf16)s[i];
}

// ---------------------------------------------------------------- WMMA GEMM
// C[M,N] = A[M,K] * W[N,K]^T + bias ; one 16x16 tile per wave, K-loop of x32 steps.
__global__ void wmma_gemm_bf16(const __bf16* __restrict__ A, const __bf16* __restrict__ W,
                               const float* __restrict__ bias,
                               __bf16* __restrict__ Cb, float* __restrict__ Cf,
                               int M, int N, int K)
{
    const int lane = threadIdx.x & 31;
    const int wid  = blockIdx.x * (blockDim.x >> 5) + (threadIdx.x >> 5);
    const int ntn  = N >> 4;
    if (wid >= (M >> 4) * ntn) return;              // uniform per-wave exit
    const int tm = (wid / ntn) << 4;
    const int tn = (wid % ntn) << 4;
    const int i  = lane & 15;                        // A row / B col / bias col
    const int h  = lane >> 4;
    const int j0 = h << 4;                           // A K-chunk start within 32

    v8f acc = {0.f,0.f,0.f,0.f,0.f,0.f,0.f,0.f};
    const __bf16* arow = A + (size_t)(tm + i) * K;
    const __bf16* brow = W + (size_t)(tn + i) * K + (h << 4);
    for (int kk = 0; kk < K; kk += 32) {
        v8bf a0 = *(const v8bf*)(arow + kk + j0);
        v8bf a1 = *(const v8bf*)(arow + kk + j0 + 8);
        v16bf a = __builtin_shufflevector(a0, a1, 0,1,2,3,4,5,6,7,8,9,10,11,12,13,14,15);
        v16bf b = *(const v16bf*)(brow + kk);
        __builtin_prefetch(arow + kk + 512, 0, 1);   // gfx1250 global_prefetch_b8
        acc = __builtin_amdgcn_wmma_f32_16x16x32_bf16(false, a, false, b,
                                                      (short)0, acc, false, false);
    }
    const float bn = bias[tn + i];
    #pragma unroll
    for (int r = 0; r < 8; ++r) {
        const int row = tm + r + (h << 3);
        const float val = acc[r] + bn;
        if (Cb) Cb[(size_t)row * N + tn + i] = (__bf16)val;
        else    Cf[(size_t)row * N + tn + i] = val;
    }
}

// ------------------------------------------- depthwise k=3 causal conv + gate
// X (B*L, 3D) bf16 -> v = conv(ch2)*conv(ch0) and x2 = conv(ch1), both (B,D,L)
__global__ void shortconv_gate(const __bf16* __restrict__ X,
                               const float* __restrict__ cw, const float* __restrict__ cb,
                               __bf16* __restrict__ vout, __bf16* __restrict__ x2out)
{
    __shared__ __bf16 tile[3][34][33];
    const int tx = threadIdx.x, ty = threadIdx.y;
    const int l0 = blockIdx.x << 5, d0 = blockIdx.y << 5, b = blockIdx.z;
    for (int r = ty; r < 34; r += 8) {
        const int l = l0 - 2 + r;
        #pragma unroll
        for (int ch = 0; ch < 3; ++ch) {
            __bf16 xv = (__bf16)0.f;
            if (l >= 0) xv = X[(size_t)(b * Lq + l) * TD + ch * Dm + d0 + tx];
            tile[ch][r][tx] = xv;
        }
    }
    __syncthreads();
    for (int dd = ty; dd < 32; dd += 8) {
        const int d = d0 + dd;
        float conv[3];
        #pragma unroll
        for (int ch = 0; ch < 3; ++ch) {
            const int o = ch * Dm + d;
            conv[ch] = cb[o] + cw[o*3+0] * (float)tile[ch][tx+0][dd]
                             + cw[o*3+1] * (float)tile[ch][tx+1][dd]
                             + cw[o*3+2] * (float)tile[ch][tx+2][dd];
        }
        const size_t oidx = (size_t)(b * Dm + d) * Lq + l0 + tx;
        vout[oidx]  = (__bf16)(conv[2] * conv[0]);   // v * x1
        x2out[oidx] = (__bf16)conv[1];
    }
}

// ------------------------------------------- implicit filter MLP + modulation
// writes k reversed+padded: krev[d, y] = k[d, L+31-y] for y in [32, L+32), zeros elsewhere
__global__ void hyena_filter(const float* __restrict__ z,
                             const float* __restrict__ fw1, const float* __restrict__ fb1,
                             const float* __restrict__ fw2, const float* __restrict__ fb2,
                             const float* __restrict__ fw3, const float* __restrict__ fb3,
                             const float* __restrict__ fw4, const float* __restrict__ freq,
                             __bf16* __restrict__ krev)
{
    __shared__ float hl[256][17];
    const int tid = threadIdx.x;
    const int l = blockIdx.x * 256 + tid;
    const float z0 = z[l*3+0], z1 = z[l*3+1], z2 = z[l*3+2];
    float h1[ORDER], h2[ORDER];
    #pragma unroll
    for (int o = 0; o < ORDER; ++o)
        h1[o] = __sinf(freq[o] * (fw1[o*3]*z0 + fw1[o*3+1]*z1 + fw1[o*3+2]*z2 + fb1[o]));
    #pragma unroll
    for (int o = 0; o < ORDER; ++o) {
        float s = fb2[o];
        #pragma unroll
        for (int e = 0; e < ORDER; ++e) s += fw2[o*ORDER+e] * h1[e];
        h2[o] = __sinf(freq[o] * s);
    }
    #pragma unroll
    for (int o = 0; o < ORDER; ++o) {
        float s = fb3[o];
        #pragma unroll
        for (int e = 0; e < ORDER; ++e) s += fw3[o*ORDER+e] * h2[e];
        hl[tid][o] = __sinf(freq[o] * s);
    }
    __syncthreads();
    const float tpos = (float)l / (float)(Lq - 1);
    const float mind = logf(1e-2f) / 1.5f, maxd = logf(1e-2f) / 0.3f;
    for (int d = 0; d < Dm; ++d) {
        float s = 0.f;
        #pragma unroll
        for (int e = 0; e < ORDER; ++e) s += hl[tid][e] * fw4[d*ORDER+e];
        const float delta = fabsf(mind + (maxd - mind) * ((float)d / (float)(Dm - 1)));
        s *= __expf(-tpos * delta);
        krev[(size_t)d * KREV_L + (Lq + 31 - l)] = (__bf16)s;   // coalesced (reversed)
    }
}

#if USE_ASYNC_G2L
// Wave-wide async copy: one b128 per lane per issue (512 B / wave / instruction).
// LDS operand = low 32 bits of the generic pointer (== LDS byte address per ISA);
// global address in a VGPR pair. Tracked by ASYNCcnt.
__device__ __forceinline__ void async_g2l_b128(unsigned ldsAddr, unsigned long long gAddr) {
    asm volatile("global_load_async_to_lds_b128 %0, %1, off"
                 :: "v"(ldsAddr), "v"(gAddr) : "memory");
}
#endif

// ------------------------------------------- long causal conv as Toeplitz-WMMA
// One wave per (b,d): y[t] = sum_l k[l] v[t-l]; A[i,j]=k[32c+i-j] (shared by all
// tiles at a given c), B[j,n]=v[256p-32c+16n+j]; acc[p] covers t in [256p,256p+256)
__global__ void __launch_bounds__(32) long_conv_wmma(
    const __bf16* __restrict__ vbf, const __bf16* __restrict__ krev,
    const __bf16* __restrict__ x2bf, const float* __restrict__ fbias,
    __bf16* __restrict__ ygd)
{
    __shared__ __align__(16) __bf16 vlds[256 + Lq + 32];   // 256-zero front margin
    __shared__ __align__(16) __bf16 klds[KREV_L];
    const int lane = threadIdx.x;
    const int d = blockIdx.x, b = blockIdx.y;
    const size_t bd = (size_t)b * Dm + d;

    { // zero v margins (DS stores, disjoint from async-filled region)
        uint4 zz = {0u,0u,0u,0u};
        if (lane < 32) ((uint4*)vlds)[lane] = zz;
        if (lane < 4)  ((uint4*)(vlds + 256 + Lq))[lane] = zz;
    }
#if USE_ASYNC_G2L
    { // stage v (8192 B) and reversed k (8320 B) via async global->LDS DMA
        const unsigned lv = (unsigned)(size_t)(void*)(vlds + 256) + (unsigned)lane * 16u;
        const unsigned long long gv =
            (unsigned long long)(size_t)(const void*)(vbf + bd * Lq) + (unsigned long long)(lane * 16);
        #pragma unroll
        for (int o = 0; o < Lq * 2; o += 512) async_g2l_b128(lv + o, gv + o);       // 16 issues
        const unsigned lk = (unsigned)(size_t)(void*)klds + (unsigned)lane * 16u;
        const unsigned long long gk =
            (unsigned long long)(size_t)(const void*)(krev + (size_t)d * KREV_L) + (unsigned long long)(lane * 16);
        #pragma unroll
        for (int o = 0; o < 8192; o += 512) async_g2l_b128(lk + o, gk + o);         // 16 issues
        if (lane < 8) async_g2l_b128(lk + 8192, gk + 8192);                         // 128 B tail
        asm volatile("s_wait_asynccnt 0x0" ::: "memory");
    }
#else
    { // fallback: VGPR round-trip staging
        const uint4* vs = (const uint4*)(vbf + bd * Lq);
        uint4* vd = (uint4*)(vlds + 256);
        for (int idx = lane; idx < Lq / 8; idx += 32) vd[idx] = vs[idx];
        const uint4* ks = (const uint4*)(krev + (size_t)d * KREV_L);
        uint4* kd = (uint4*)klds;
        for (int idx = lane; idx < KREV_L / 8; idx += 32) kd[idx] = ks[idx];
    }
#endif
    __syncthreads();

    const int i  = lane & 15;        // A row / B col
    const int h  = lane >> 4;
    const int j0 = h << 4;
    v8f acc[16];
    const v8f zv = {0.f,0.f,0.f,0.f,0.f,0.f,0.f,0.f};
    #pragma unroll
    for (int p = 0; p < 16; ++p) acc[p] = zv;

    for (int c = 0; c <= 128; ++c) {
        const int y0 = (Lq + 31) - 32 * c - i;       // krev[y0+j] == k[32c+i-j] (0-padded)
        v16bf a;
        #pragma unroll
        for (int e = 0; e < 8; ++e) {
            a[e]     = klds[y0 + j0 + e];
            a[e + 8] = klds[y0 + j0 + 8 + e];
        }
        const int pmin = (c <= 8) ? 0 : ((c - 1) >> 3);     // causality cut
        const int vb0  = 256 - 32 * c + 16 * i + (h << 4);  // margin-offset v base
        #pragma unroll
        for (int p = 0; p < 16; ++p) {
            if (p < pmin) continue;                         // uniform scalar branch
            v16bf bb = *(const v16bf*)(vlds + vb0 + (p << 8));
            acc[p] = __builtin_amdgcn_wmma_f32_16x16x32_bf16(false, a, false, bb,
                                                             (short)0, acc[p], false, false);
        }
    }

    const float fb = fbias[d];
    #pragma unroll
    for (int p = 0; p < 16; ++p) {
        const int tb = (p << 8) + 16 * i + (h << 3);        // t = 256p + 16n + M
        #pragma unroll
        for (int r = 0; r < 8; ++r) {
            const int t = tb + r;
            const float yv = acc[p][r] + (float)vlds[256 + t] * fb;  // + v*filter_bias
            const float g  = (float)x2bf[bd * Lq + t];               // post-gate
            ygd[bd * Lq + t] = (__bf16)(yv * g);
        }
    }
}

// ------------------------------------------- (B,D,L) -> (B*L, D) transpose
__global__ void transpose_dl_to_ld(const __bf16* __restrict__ src, __bf16* __restrict__ dst)
{
    __shared__ __bf16 tile[32][33];
    const int tx = threadIdx.x, ty = threadIdx.y;
    const int l0 = blockIdx.x << 5, d0 = blockIdx.y << 5, b = blockIdx.z;
    for (int r = ty; r < 32; r += 8)
        tile[r][tx] = src[(size_t)(b * Dm + d0 + r) * Lq + l0 + tx];
    __syncthreads();
    for (int r = ty; r < 32; r += 8)
        dst[(size_t)(b * Lq + l0 + r) * Dm + d0 + tx] = tile[tx][r];
}

// ---------------------------------------------------------------- launch
extern "C" void kernel_launch(void* const* d_in, const int* in_sizes, int n_in,
                              void* d_out, int out_size, void* d_ws, size_t ws_size,
                              hipStream_t stream)
{
    const float* u      = (const float*)d_in[0];
    const float* w_in   = (const float*)d_in[1];
    const float* b_in   = (const float*)d_in[2];
    const float* w_out  = (const float*)d_in[3];
    const float* b_out  = (const float*)d_in[4];
    const float* conv_w = (const float*)d_in[5];
    const float* conv_b = (const float*)d_in[6];
    const float* z      = (const float*)d_in[7];
    const float* fw1    = (const float*)d_in[8];
    const float* fb1    = (const float*)d_in[9];
    const float* fw2    = (const float*)d_in[10];
    const float* fb2    = (const float*)d_in[11];
    const float* fw3    = (const float*)d_in[12];
    const float* fb3    = (const float*)d_in[13];
    const float* fw4    = (const float*)d_in[14];
    const float* freq   = (const float*)d_in[15];
    const float* fbias  = (const float*)d_in[16];
    (void)in_sizes; (void)n_in; (void)out_size; (void)ws_size;

    char* ws = (char*)d_ws;
    size_t off = 0;
    auto carve = [&](size_t bytes) {
        char* p = ws + off;
        off = (off + bytes + 255) & ~(size_t)255;
        return p;
    };
    __bf16* ubf  = (__bf16*)carve((size_t)BL * Dm * 2);        // 24 MB
    __bf16* wibf = (__bf16*)carve((size_t)TD * Dm * 2);
    __bf16* wobf = (__bf16*)carve((size_t)Dm * Dm * 2);
    __bf16* X    = (__bf16*)carve((size_t)BL * TD * 2);        // 72 MB
    __bf16* vbf  = (__bf16*)carve((size_t)Bsz * Dm * Lq * 2);
    __bf16* x2bf = (__bf16*)carve((size_t)Bsz * Dm * Lq * 2);
    __bf16* krev = (__bf16*)carve((size_t)Dm * KREV_L * 2);
    __bf16* ygd  = (__bf16*)carve((size_t)Bsz * Dm * Lq * 2);
    __bf16* ygt  = (__bf16*)carve((size_t)Bsz * Lq * Dm * 2);  // total ~203 MB

    hipMemsetAsync(krev, 0, (size_t)Dm * KREV_L * 2, stream);  // zero pad margins

    { int n = BL * Dm; cvt_f32_to_bf16<<<(n + 255) / 256, 256, 0, stream>>>(u, ubf, n); }
    { int n = TD * Dm; cvt_f32_to_bf16<<<(n + 255) / 256, 256, 0, stream>>>(w_in, wibf, n); }
    { int n = Dm * Dm; cvt_f32_to_bf16<<<(n + 255) / 256, 256, 0, stream>>>(w_out, wobf, n); }

    // in-projection: X = u @ w_in^T + b_in
    { int total = (BL / 16) * (TD / 16);
      wmma_gemm_bf16<<<(total + 7) / 8, 256, 0, stream>>>(ubf, wibf, b_in, X, nullptr, BL, TD, Dm); }

    shortconv_gate<<<dim3(Lq / 32, Dm / 32, Bsz), dim3(32, 8), 0, stream>>>(X, conv_w, conv_b, vbf, x2bf);

    hyena_filter<<<Lq / 256, 256, 0, stream>>>(z, fw1, fb1, fw2, fb2, fw3, fb3, fw4, freq, krev);

    long_conv_wmma<<<dim3(Dm, Bsz), 32, 0, stream>>>(vbf, krev, x2bf, fbias, ygd);

    transpose_dl_to_ld<<<dim3(Lq / 32, Dm / 32, Bsz), dim3(32, 8), 0, stream>>>(ygd, ygt);

    // out-projection: out = yg @ w_out^T + b_out  (fp32 output)
    { int total = (BL / 16) * (Dm / 16);
      wmma_gemm_bf16<<<(total + 7) / 8, 256, 0, stream>>>(ygt, wobf, b_out, nullptr, (float*)d_out, BL, Dm, Dm); }
}
